// IEL_23562190586436
// MI455X (gfx1250) — matrix operand order repaired and checked by reading, not compiled, not run on hardware
//
// MI455X (gfx1250, wave32) implementation of a VMamba SS2D block.
//
// Pipeline: conv1x1(WMMA) -> dwconv3x3 -> channel-LN -> proj(WMMA,680x170)
//        -> dwconv3x3+silu -> LDS transpose -> 4x dir-proj(WMMA,43x340)
//        -> wave-chunked selective scan (shfl affine scan, fwd/bwd dirs)
//        -> combine+LN+silu -> out-proj(WMMA,170x340) -> gated branch
//        -> conv1x1(WMMA, 64x170).
//
// GEMM kernels: weights staged to LDS as f16 (ds_load_b128 fragment reads),
// activations read by per-element guarded scalar loads, f32 WMMA accumulate.
// Flip-free scan: k=2,3 iterate positions in reverse and store results at the
// position they were read from, which fuses flip + unflip. k=1,3 run on the
// HW-transposed sequence; the final combine reads those at lcol = w*H + h.

#include <hip/hip_runtime.h>
#include <hip/hip_bf16.h>

typedef __attribute__((ext_vector_type(16))) _Float16 v16h;
typedef __attribute__((ext_vector_type(8)))  _Float16 v8h;
typedef __attribute__((ext_vector_type(8)))  float    v8f;

#define LL   4096
#define HH   64
#define WW   64
#define BB   2
#define DINC 340
#define HFC  170
#define NST  16
#define RNK  11
#define KPAD 352   // max padded K across all GEMMs (340 -> 352)

__device__ __forceinline__ float softplusf(float x) {
  return x > 20.f ? x : log1pf(__expf(x));
}
__device__ __forceinline__ float siluf(float x) {
  return x / (1.f + __expf(-x));
}

// ---------------------------------------------------------------------------
// B fragment: 32x16 f16; lane = n + 16*g, elems e<8 -> K = k0+8g+e,
// e>=8 -> K = k0+16+8g+(e-8) (per-element guarded scalar loads).
// ---------------------------------------------------------------------------
__device__ __forceinline__ v16h load_b_frag(const float* __restrict__ X,
                                            int Kd, int ld, int k0, int n0, int lane) {
  v16h b = {};
  int g = lane >> 4;
  const float* col = X + n0 + (lane & 15);
#pragma unroll
  for (int e = 0; e < 8; ++e) {
    int K0 = k0 + 8 * g + e;
    int K1 = k0 + 16 + 8 * g + e;
    if (K0 < Kd) b[e] = (_Float16)col[(size_t)K0 * ld];
    if (K1 < Kd) b[8 + e] = (_Float16)col[(size_t)K1 * ld];
  }
  return b;
}

// Y[M,Nlen] = Wm[M,Kd] * X[Kd,Nlen], batched over blockIdx.z.
// Block: 2 waves sharing one 16-row weight tile (LDS f16, zero-padded).
// Each wave computes a 16(M) x 64(N) tile via 4 WMMAs per 32-wide K step.
// A fragment layout: lane = m + 16*g; elems e<8 -> K = k0+8g+e,
// e>=8 -> K = k0+16+8g+(e-8): two aligned 16B LDS reads per step.
__global__ void k_gemm_wmma(const float* __restrict__ Wm, const float* __restrict__ X,
                            float* __restrict__ Y, int M, int Kd, int Nlen,
                            size_t strideW, size_t strideX, size_t strideY) {
  __shared__ _Float16 As[16 * KPAD];

  int lane = threadIdx.x;
  int wid  = threadIdx.y;
  int tid  = wid * 32 + lane;
  int m0 = blockIdx.y * 16;
  int n0 = (blockIdx.x * 2 + wid) * 64;
  const float* Wb = Wm + strideW * blockIdx.z;
  const float* Xb = X  + strideX * blockIdx.z;
  float*       Yb = Y  + strideY * blockIdx.z;

  int Kpad = (Kd + 31) & ~31;
  for (int m = 0; m < 16; ++m) {
    for (int k = tid; k < Kpad; k += 64) {
      float v = ((m0 + m) < M && k < Kd) ? Wb[(size_t)(m0 + m) * Kd + k] : 0.f;
      As[m * KPAD + k] = (_Float16)v;
    }
  }
  __syncthreads();

  int mrow = lane & 15, g = lane >> 4;
  const _Float16* arow = &As[mrow * KPAD];

  v8f c0 = {}, c1 = {}, c2 = {}, c3 = {};
  for (int k0 = 0; k0 < Kd; k0 += 32) {
    // prefetch next K panel of activations (global_prefetch_b8)
    if (k0 + 32 < Kd) {
      int Kp = k0 + 32 + lane;
      if (Kp < Kd) __builtin_prefetch(Xb + (size_t)Kp * Nlen + n0, 0, 1);
    }
    v8h lo = *(const v8h*)(arow + k0 + 8 * g);        // ds_load_b128
    v8h hi = *(const v8h*)(arow + k0 + 16 + 8 * g);   // ds_load_b128
    v16h a = {};
#pragma unroll
    for (int e = 0; e < 8; ++e) { a[e] = lo[e]; a[8 + e] = hi[e]; }

    v16h b0 = load_b_frag(Xb, Kd, Nlen, k0, n0 +  0, lane);
    v16h b1 = load_b_frag(Xb, Kd, Nlen, k0, n0 + 16, lane);
    v16h b2 = load_b_frag(Xb, Kd, Nlen, k0, n0 + 32, lane);
    v16h b3 = load_b_frag(Xb, Kd, Nlen, k0, n0 + 48, lane);
    c0 = __builtin_amdgcn_wmma_f32_16x16x32_f16(false, a, false, b0, (short)0, c0, false, false);
    c1 = __builtin_amdgcn_wmma_f32_16x16x32_f16(false, a, false, b1, (short)0, c1, false, false);
    c2 = __builtin_amdgcn_wmma_f32_16x16x32_f16(false, a, false, b2, (short)0, c2, false, false);
    c3 = __builtin_amdgcn_wmma_f32_16x16x32_f16(false, a, false, b3, (short)0, c3, false, false);
  }

  int n = lane & 15;
#pragma unroll
  for (int v = 0; v < 8; ++v) {
    int m = m0 + v + 8 * g;
    if (m < M) {
      float* yr = Yb + (size_t)m * Nlen + n0;
      yr[n +  0] = c0[v];
      yr[n + 16] = c1[v];
      yr[n + 32] = c2[v];
      yr[n + 48] = c3[v];
    }
  }
}

// ---------------------------------------------------------------------------
// Depthwise 3x3 conv, SAME padding, optional bias + silu.
// grid: (LL/256, C, B); channel-major (c-row stride LL inside a batch block).
// ---------------------------------------------------------------------------
__global__ void k_dwconv(const float* __restrict__ in, const float* __restrict__ wgt,
                         const float* __restrict__ bias, float* __restrict__ out,
                         size_t bsIn, size_t bsOut, int silu) {
  int l = blockIdx.x * blockDim.x + threadIdx.x;
  int c = blockIdx.y, b = blockIdx.z;
  int h = l >> 6, w = l & 63;
  const float* src = in + (size_t)b * bsIn + (size_t)c * LL;
  float acc = bias ? bias[c] : 0.f;
#pragma unroll
  for (int kh = 0; kh < 3; ++kh) {
    int hh = h + kh - 1;
    if ((unsigned)hh > 63u) continue;
#pragma unroll
    for (int kw = 0; kw < 3; ++kw) {
      int ww = w + kw - 1;
      if ((unsigned)ww > 63u) continue;
      acc += src[(hh << 6) + ww] * wgt[c * 9 + kh * 3 + kw];
    }
  }
  if (silu) acc = siluf(acc);
  out[(size_t)b * bsOut + (size_t)c * LL + l] = acc;
}

// ---------------------------------------------------------------------------
// Channel LayerNorm: per pixel over C channels (channel-major layout).
// ---------------------------------------------------------------------------
__global__ void k_ln(const float* __restrict__ in, float* __restrict__ out,
                     const float* __restrict__ g, const float* __restrict__ b,
                     int C, size_t bsIn, size_t bsOut, float eps) {
  int idx = blockIdx.x * blockDim.x + threadIdx.x;   // over B*L
  int bb = idx / LL, l = idx % LL;
  const float* x = in + (size_t)bb * bsIn + l;
  float s = 0.f, s2 = 0.f;
  for (int c = 0; c < C; ++c) {
    float v = x[(size_t)c * LL];
    s += v; s2 += v * v;
  }
  float m = s / C;
  float inv = rsqrtf(s2 / C - m * m + eps);
  float* y = out + (size_t)bb * bsOut + l;
  for (int c = 0; c < C; ++c) {
    float v = x[(size_t)c * LL];
    y[(size_t)c * LL] = (v - m) * inv * g[c] + b[c];
  }
}

// ---------------------------------------------------------------------------
// 64x64 plane transpose via LDS (seq1[w*H+h] = seq0[h*W+w]).
// ---------------------------------------------------------------------------
__global__ void k_transpose64(const float* __restrict__ src, float* __restrict__ dst) {
  __shared__ float tile[32][33];
  int plane = blockIdx.z;
  const float* s = src + (size_t)plane * LL;
  float*       d = dst + (size_t)plane * LL;
  int tx = threadIdx.x, ty = threadIdx.y;
  int h0 = blockIdx.y * 32, w0 = blockIdx.x * 32;
  for (int i = ty; i < 32; i += 8)
    tile[i][tx] = s[(h0 + i) * 64 + (w0 + tx)];
  __syncthreads();
  for (int i = ty; i < 32; i += 8)
    d[(w0 + i) * 64 + (h0 + tx)] = tile[tx][i];
}

// ---------------------------------------------------------------------------
// Selective scan: one wave per (b,k,d). Lane owns 128 contiguous steps.
// Pass 1: compose affine (aP, cA) per state. Wave shfl_up scan -> h_start.
// Pass 2: replay with correct h, emit y (+ u*D). dt recomputed from xdbl rows
// 0..10 (R), B rows 11..26, C rows 27..42 (48-row padded layout per (b,k)).
// Directions k>=2 scan backwards; outputs stored at the read position.
// ---------------------------------------------------------------------------
__global__ void k_scan(const float* __restrict__ seq0, const float* __restrict__ seq1,
                       const float* __restrict__ xdbl, const float* __restrict__ dtw_all,
                       const float* __restrict__ dtb_all, const float* __restrict__ Alog_all,
                       const float* __restrict__ Ds_all, float* __restrict__ yall) {
  int wid  = threadIdx.x >> 5;
  int lane = threadIdx.x & 31;
  int gw = blockIdx.x * (blockDim.x >> 5) + wid;  // 0 .. B*4*DIN-1
  int d  = gw % DINC;
  int bk = gw / DINC;
  int k  = bk % 4;
  int b  = bk / 4;

  const float* seq   = ((k & 1) ? seq1 : seq0) + (size_t)b * DINC * LL + (size_t)d * LL;
  const float* xd_bk = xdbl + (size_t)(b * 4 + k) * 48 * LL;
  const float* dtrow = dtw_all + (size_t)(k * DINC + d) * RNK;
  float dtb = dtb_all[k * DINC + d];
  const float* Arow  = Alog_all + (size_t)(k * DINC + d) * NST;
  float Dv = Ds_all[k * DINC + d];
  float* yo = yall + ((size_t)(k * BB + b) * DINC + d) * LL;
  int dir = (k >= 2);

  float dtw_r[RNK];
#pragma unroll
  for (int r = 0; r < RNK; ++r) dtw_r[r] = dtrow[r];
  float A[NST];
#pragma unroll
  for (int n = 0; n < NST; ++n) A[n] = -__expf(Arow[n]);

  const int CH = LL / 32;       // 128 steps per lane
  int jbase = lane * CH;

  float aP[NST], cA[NST];
#pragma unroll
  for (int n = 0; n < NST; ++n) { aP[n] = 1.f; cA[n] = 0.f; }

  for (int i = 0; i < CH; ++i) {
    int j = jbase + i;
    int pos = dir ? (LL - 1 - j) : j;
    float acc = dtb;
#pragma unroll
    for (int r = 0; r < RNK; ++r) acc += dtw_r[r] * xd_bk[(size_t)r * LL + pos];
    float dt = softplusf(acc);
    float du = dt * seq[pos];
#pragma unroll
    for (int n = 0; n < NST; ++n) {
      float a = __expf(dt * A[n]);
      float x = du * xd_bk[(size_t)(RNK + n) * LL + pos];
      cA[n] = a * cA[n] + x;
      aP[n] = a * aP[n];
    }
  }

  // inclusive affine scan across lanes (Hillis-Steele over composition)
#pragma unroll
  for (int off = 1; off < 32; off <<= 1) {
#pragma unroll
    for (int n = 0; n < NST; ++n) {
      float al = __shfl_up(aP[n], off, 32);
      float cl = __shfl_up(cA[n], off, 32);
      if (lane >= off) {
        cA[n] = aP[n] * cl + cA[n];
        aP[n] = aP[n] * al;
      }
    }
  }
  float h[NST];
#pragma unroll
  for (int n = 0; n < NST; ++n) {
    float cex = __shfl_up(cA[n], 1, 32);   // exclusive prefix, h0 = 0
    h[n] = (lane == 0) ? 0.f : cex;
  }

  for (int i = 0; i < CH; ++i) {
    int j = jbase + i;
    int pos = dir ? (LL - 1 - j) : j;
    float acc = dtb;
#pragma unroll
    for (int r = 0; r < RNK; ++r) acc += dtw_r[r] * xd_bk[(size_t)r * LL + pos];
    float dt = softplusf(acc);
    float u = seq[pos];
    float du = dt * u;
    float y = u * Dv;
#pragma unroll
    for (int n = 0; n < NST; ++n) {
      float a = __expf(dt * A[n]);
      float x = du * xd_bk[(size_t)(RNK + n) * LL + pos];
      h[n] = a * h[n] + x;
      y += h[n] * xd_bk[(size_t)(RNK + NST + n) * LL + pos];
    }
    yo[pos] = y;
  }
}

// ---------------------------------------------------------------------------
// Combine 4 directions + out-LN + silu(z) gate. Channel-major output.
// ---------------------------------------------------------------------------
__global__ void k_combine(const float* __restrict__ yall, const float* __restrict__ xz,
                          const float* __restrict__ g, const float* __restrict__ bvec,
                          float* __restrict__ ysil) {
  int idx = blockIdx.x * blockDim.x + threadIdx.x;   // over B*L
  int b = idx / LL, l = idx % LL;
  int h = l >> 6, w = l & 63;
  int lcol = (w << 6) | h;
  const size_t PL = (size_t)DINC * LL;
  const float* y0 = yall + (size_t)(0 * BB + b) * PL;
  const float* y1 = yall + (size_t)(1 * BB + b) * PL;
  const float* y2 = yall + (size_t)(2 * BB + b) * PL;
  const float* y3 = yall + (size_t)(3 * BB + b) * PL;
  const float* zp = xz + (size_t)b * 680 * LL + (size_t)DINC * LL;   // z rows

  float s = 0.f, s2 = 0.f;
  for (int d = 0; d < DINC; ++d) {
    float v = y0[(size_t)d * LL + l] + y2[(size_t)d * LL + l]
            + y1[(size_t)d * LL + lcol] + y3[(size_t)d * LL + lcol];
    s += v; s2 += v * v;
  }
  float m = s / DINC;
  float inv = rsqrtf(s2 / DINC - m * m + 1e-5f);
  float* out = ysil + (size_t)b * PL;
  for (int d = 0; d < DINC; ++d) {
    float v = y0[(size_t)d * LL + l] + y2[(size_t)d * LL + l]
            + y1[(size_t)d * LL + lcol] + y3[(size_t)d * LL + lcol];
    float yn = (v - m) * inv * g[d] + bvec[d];
    float zz = zp[(size_t)d * LL + l];
    out[(size_t)d * LL + l] = yn * siluf(zz);
  }
}

// ---------------------------------------------------------------------------
// Gated residual branch: x1f = (tanh(r)+x1) * (tanh(dwconv(x2,w_dw2))+x2).
// ---------------------------------------------------------------------------
__global__ void k_branch(const float* __restrict__ xdv, const float* __restrict__ yss,
                         const float* __restrict__ w_dw2, float* __restrict__ x1f) {
  int l = blockIdx.x * blockDim.x + threadIdx.x;
  int c = blockIdx.y, b = blockIdx.z;
  int h = l >> 6, w = l & 63;
  const float* x2r = xdv + (size_t)b * DINC * LL + (size_t)(HFC + c) * LL;
  float acc = 0.f;
#pragma unroll
  for (int kh = 0; kh < 3; ++kh) {
    int hh = h + kh - 1;
    if ((unsigned)hh > 63u) continue;
#pragma unroll
    for (int kw = 0; kw < 3; ++kw) {
      int ww = w + kw - 1;
      if ((unsigned)ww > 63u) continue;
      acc += x2r[(hh << 6) + ww] * w_dw2[c * 9 + kh * 3 + kw];
    }
  }
  float x2v = tanhf(acc) + x2r[l];
  float rv  = yss[(size_t)b * HFC * LL + (size_t)c * LL + l];
  float x1v = tanhf(rv) + xdv[(size_t)b * DINC * LL + (size_t)c * LL + l];
  x1f[(size_t)b * HFC * LL + (size_t)c * LL + l] = x1v * x2v;
}

// ---------------------------------------------------------------------------
extern "C" void kernel_launch(void* const* d_in, const int* in_sizes, int n_in,
                              void* d_out, int out_size, void* d_ws, size_t ws_size,
                              hipStream_t stream) {
  (void)in_sizes; (void)n_in; (void)out_size; (void)ws_size;
  const float* x          = (const float*)d_in[0];
  const float* w_in       = (const float*)d_in[1];
  const float* w_dw       = (const float*)d_in[2];
  const float* w_dw2      = (const float*)d_in[3];
  const float* ln1_g      = (const float*)d_in[4];
  const float* ln1_b      = (const float*)d_in[5];
  const float* ssm_in_w   = (const float*)d_in[6];
  const float* ssm_conv_w = (const float*)d_in[7];
  const float* ssm_conv_b = (const float*)d_in[8];
  const float* x_proj_w   = (const float*)d_in[9];
  const float* dt_w       = (const float*)d_in[10];
  const float* dt_b       = (const float*)d_in[11];
  const float* A_logs     = (const float*)d_in[12];
  const float* Ds         = (const float*)d_in[13];
  const float* ong        = (const float*)d_in[14];
  const float* onb        = (const float*)d_in[15];
  const float* ssm_out_w  = (const float*)d_in[16];
  const float* w_out      = (const float*)d_in[17];
  float* out = (float*)d_out;

  const size_t L = LL;
  float* ws = (float*)d_ws;
  size_t off = 0;
  auto alloc = [&](size_t n) { float* p = ws + off; off += n; return p; };
  float* xp   = alloc((size_t)BB * DINC * L);   // conv1x1 in
  float* xdv  = alloc((size_t)BB * DINC * L);   // dwconv (x1|x2 halves)
  float* rr   = alloc((size_t)BB * HFC * L);    // LN(x1)
  float* xz   = alloc((size_t)BB * 680 * L);    // [xc | z]
  float* seq0 = alloc((size_t)BB * DINC * L);   // silu(dwconv(xc))
  float* seq1 = alloc((size_t)BB * DINC * L);   // HW-transposed
  float* xdbl = alloc((size_t)BB * 4 * 48 * L); // per-dir proj (rows padded 43->48)
  float* yall = alloc((size_t)4 * BB * DINC * L);
  float* ysil = alloc((size_t)BB * DINC * L);
  float* yss  = alloc((size_t)BB * HFC * L);
  float* x1f  = alloc((size_t)BB * HFC * L);

  dim3 gblk(32, 2, 1);  // 2 waves per block, each a 16x64 WMMA tile

  // 1: xp = w_in(340x64) @ x
  k_gemm_wmma<<<dim3(L / 128, 22, BB), gblk, 0, stream>>>(
      w_in, x, xp, DINC, 64, (int)L, 0, (size_t)64 * L, (size_t)DINC * L);
  // 2: xd = dwconv3x3(xp)
  k_dwconv<<<dim3(L / 256, DINC, BB), 256, 0, stream>>>(
      xp, w_dw, nullptr, xdv, (size_t)DINC * L, (size_t)DINC * L, 0);
  // 3: r = LN(x1) over 170 channels
  k_ln<<<dim3((BB * L) / 256), 256, 0, stream>>>(
      xdv, rr, ln1_g, ln1_b, HFC, (size_t)DINC * L, (size_t)HFC * L, 1e-6f);
  // 4: xz = ssm_in_w(680x170) @ r
  k_gemm_wmma<<<dim3(L / 128, 43, BB), gblk, 0, stream>>>(
      ssm_in_w, rr, xz, 680, HFC, (int)L, 0, (size_t)HFC * L, (size_t)680 * L);
  // 5: seq0 = silu(dwconv3x3(xc) + b)
  k_dwconv<<<dim3(L / 256, DINC, BB), 256, 0, stream>>>(
      xz, ssm_conv_w, ssm_conv_b, seq0, (size_t)680 * L, (size_t)DINC * L, 1);
  // 6: seq1 = transpose_HW(seq0)  (LDS tiles)
  k_transpose64<<<dim3(2, 2, BB * DINC), dim3(32, 8), 0, stream>>>(seq0, seq1);
  // 7: per-direction projections xdbl[k] = x_proj_w[k](43x340) @ seq_{k%2}
  for (int k = 0; k < 4; ++k) {
    const float* sk = (k & 1) ? seq1 : seq0;
    k_gemm_wmma<<<dim3(L / 128, 3, BB), gblk, 0, stream>>>(
        x_proj_w + (size_t)k * 43 * DINC, sk, xdbl + (size_t)k * 48 * L,
        43, DINC, (int)L, 0, (size_t)DINC * L, (size_t)4 * 48 * L);
  }
  // 8: selective scan, one wave per (b,k,d): 2720 waves
  k_scan<<<dim3((BB * 4 * DINC) / 8), 256, 0, stream>>>(
      seq0, seq1, xdbl, dt_w, dt_b, A_logs, Ds, yall);
  // 9: combine 4 dirs + out-LN + silu(z)
  k_combine<<<dim3((BB * L) / 256), 256, 0, stream>>>(yall, xz, ong, onb, ysil);
  // 10: yss = ssm_out_w(170x340) @ ysil
  k_gemm_wmma<<<dim3(L / 128, 11, BB), gblk, 0, stream>>>(
      ssm_out_w, ysil, yss, HFC, DINC, (int)L, 0, (size_t)DINC * L, (size_t)HFC * L);
  // 11: gated residual branch -> x1f
  k_branch<<<dim3(L / 256, HFC, BB), 256, 0, stream>>>(xdv, yss, w_dw2, x1f);
  // 12: out = w_out(64x170) @ x1f
  k_gemm_wmma<<<dim3(L / 128, 4, BB), gblk, 0, stream>>>(
      w_out, x1f, out, 64, HFC, (int)L, 0, (size_t)HFC * L, (size_t)64 * L);
}
